// Attention_12008728560179
// MI455X (gfx1250) — matrix operand order repaired and checked by reading, not compiled, not run on hardware
//
#include <hip/hip_runtime.h>
#include <stdint.h>

// ---------------------------------------------------------------------------
// Fused MHSA for MI455X (gfx1250, wave32, WMMA, async global->LDS).
// All GEMMs on v_wmma_f32_16x16x32_bf16 (f32 accumulate).
// Attention: flash-style online softmax; K/V chunks staged into LDS once per
// block via GLOBAL_LOAD_ASYNC_TO_LDS_B128 (ASYNCcnt, double-buffered) and
// shared by all 8 waves -> 8x less L2 traffic than per-wave streaming.
// ---------------------------------------------------------------------------

#define HEADS  4
#define DHEAD  64
#define HIDDEN 256          // HEADS*DHEAD
#define DIM    256
#define NTOK   4096         // 64*64
#define BATCH  2
#define QSCALE 0.125f       // DHEAD^-0.5

typedef __bf16 bf16;
typedef __attribute__((ext_vector_type(16))) __bf16 v16bf;
typedef __attribute__((ext_vector_type(8)))  __bf16 v8bf;
typedef __attribute__((ext_vector_type(8)))  float  v8f;

__device__ __forceinline__ v8f wmma_bf16(v16bf a, v16bf b, v8f c) {
  return __builtin_amdgcn_wmma_f32_16x16x32_bf16(false, a, false, b,
                                                 (short)0, c, false, false);
}

union V16 { v16bf v; v8bf h[2]; };

// B-fragment: 16 contiguous bf16 per lane (K = e + half*16, N = lane%16).
__device__ __forceinline__ v16bf ld_frag16(const bf16* p) {
  V16 u; u.h[0] = *(const v8bf*)p; u.h[1] = *(const v8bf*)(p + 8); return u.v;
}
// A-fragment: two contiguous 8-element runs (K = e + half*16).
__device__ __forceinline__ v16bf ld_frag8x2(const bf16* p0, const bf16* p1) {
  V16 u; u.h[0] = *(const v8bf*)p0; u.h[1] = *(const v8bf*)p1; return u.v;
}
// A-fragment built from fp32 weights (row-major), converted to bf16.
__device__ __forceinline__ v16bf cvt_afrag_f32(const float* p0, const float* p1) {
  v8f a = *(const v8f*)p0;
  v8f b = *(const v8f*)p1;
  v16bf r;
  #pragma unroll
  for (int i = 0; i < 8; ++i) { r[i] = (__bf16)a[i]; r[i + 8] = (__bf16)b[i]; }
  return r;
}

// Async global->LDS copy, 16B per lane. LDS dest offset = low 32 bits of the
// generic shared-pointer (flat LDS aperture maps addr[31:0] -> LDS offset).
__device__ __forceinline__ void async_copy_b128(void* lds_ptr, const void* gptr) {
  uint32_t lds_off = (uint32_t)(uintptr_t)lds_ptr;
  asm volatile("global_load_async_to_lds_b128 %0, %1, off"
               :: "v"(lds_off), "v"(gptr) : "memory");
}

// ---------------------------------------------------------------------------
// Kernel 1: qkv = w_qkv(768x256) @ x[b](256x4096).  Each wave computes a
// 64(o) x 16(n) block (4 WMMA tiles): the strided-x B-fragment is loaded once
// per K-step and reused by 4 WMMAs.  Writes bf16:
//   Q [b][h][n][d] (scaled), K [b][h][n][d], V [b][h][d][n].
// ---------------------------------------------------------------------------
__global__ __launch_bounds__(256) void qkv_proj_kernel(
    const float* __restrict__ x, const float* __restrict__ w_qkv,
    bf16* __restrict__ Q, bf16* __restrict__ K, bf16* __restrict__ V) {
  const int wave = threadIdx.x >> 5;
  const int lane = threadIdx.x & 31;
  const int half = lane >> 4;
  const int l16  = lane & 15;
  const int gid  = blockIdx.x * 8 + wave;        // 2*12*256 = 6144 waves
  const int b    = gid / (12 * 256);
  const int rem  = gid % (12 * 256);
  const int oblk = rem / 256;                    // 12 blocks of 64 o-rows
  const int nt   = rem % 256;                    // 256 n-tiles

  const float* xb = x + (size_t)b * DIM * NTOK;
  const int col   = nt * 16 + l16;

  v8f acc[4] = {{}, {}, {}, {}};
  #pragma unroll
  for (int kb = 0; kb < 8; ++kb) {
    // B fragment from x[c][n]: K-dim strided by NTOK (coalesced across lanes).
    v16bf bfrag;
    const float* bp = xb + (size_t)(kb * 32 + half * 16) * NTOK + col;
    #pragma unroll
    for (int e = 0; e < 16; ++e) bfrag[e] = (__bf16)bp[(size_t)e * NTOK];
    #pragma unroll
    for (int j = 0; j < 4; ++j) {
      const int arow = oblk * 64 + j * 16 + l16;
      const float* ap = w_qkv + (size_t)arow * DIM + kb * 32 + half * 16;
      acc[j] = wmma_bf16(cvt_afrag_f32(ap, ap + 8), bfrag, acc[j]);
    }
  }

  // D layout: tile j, element r -> o = oblk*64 + j*16 + r + half*8.
  // One 64-row o-block == exactly one head of exactly one of Q/K/V.
  const int hd = oblk & 3;                       // head within region
  #pragma unroll
  for (int j = 0; j < 4; ++j) {
    #pragma unroll
    for (int r = 0; r < 8; ++r) {
      const int d = j * 16 + r + half * 8;       // 0..63 within head
      const float val = acc[j][r];
      if (oblk < 4) {
        Q[(((size_t)b * HEADS + hd) * NTOK + col) * DHEAD + d] = (__bf16)(val * QSCALE);
      } else if (oblk < 8) {
        K[(((size_t)b * HEADS + hd) * NTOK + col) * DHEAD + d] = (__bf16)val;
      } else {
        V[(((size_t)b * HEADS + hd) * DHEAD + d) * NTOK + col] = (__bf16)val;
      }
    }
  }
}

// ---------------------------------------------------------------------------
// Kernel 2: flash attention.  One block = 8 waves = one (b, h, 128 Q rows).
// 128 iterations over 32-key chunks; each chunk's K (32x64) and V (64x32)
// tiles are async-DMA'd into LDS once (double-buffered) and consumed by all
// 8 waves.  Per wave/chunk: 4 WMMAs for S, shfl-based online softmax,
// D->A transpose of P via per-wave LDS tile, 4 WMMAs for P@V.
// ---------------------------------------------------------------------------
__global__ __launch_bounds__(256) void attn_kernel(
    const bf16* __restrict__ Q, const bf16* __restrict__ K,
    const bf16* __restrict__ V, bf16* __restrict__ O) {
  __shared__ bf16 kbuf[2][32 * 64];              // [m 0..31][d 0..63]
  __shared__ bf16 vbuf[2][64 * 32];              // [d 0..63][m 0..31]
  __shared__ bf16 pbuf[8][16 * 32];              // per-wave P tile

  const int tid  = threadIdx.x;
  const int wave = tid >> 5;
  const int lane = tid & 31;
  const int half = lane >> 4;
  const int l16  = lane & 15;
  const int blk  = blockIdx.x;                   // 2*4*32 = 256 blocks
  const int b    = blk >> 7;                     // / (4*32)
  const int h    = (blk >> 5) & 3;
  const int qt   = (blk & 31) * 8 + wave;        // this wave's 16-row Q tile

  const bf16* Qbh = Q + ((size_t)b * HEADS + h) * NTOK * DHEAD;
  const bf16* Kbh = K + ((size_t)b * HEADS + h) * NTOK * DHEAD;
  const bf16* Vbh = V + ((size_t)b * HEADS + h) * DHEAD * NTOK;

  // Per-thread source/dest slices for the cooperative K/V chunk copies:
  // K chunk: 32 rows x 128B; thread t moves 16B of row t>>3.
  const int krow = tid >> 3, koff = (tid & 7) * 8;
  // V chunk: 64 rows x 64B; thread t moves 16B of row t>>2.
  const int vrow = tid >> 2, voff = (tid & 3) * 8;

  // Prologue: stage chunk 0 into buffer 0.
  async_copy_b128(&kbuf[0][krow * 64 + koff], Kbh + (size_t)krow * DHEAD + koff);
  async_copy_b128(&vbuf[0][vrow * 32 + voff], Vbh + (size_t)vrow * NTOK + voff);

  // Q A-fragments (16x64), resident for the whole loop.
  v16bf qf[2];
  {
    const bf16* qr = Qbh + (size_t)(qt * 16 + l16) * DHEAD;
    #pragma unroll
    for (int kb = 0; kb < 2; ++kb)
      qf[kb] = ld_frag8x2(qr + kb * 32 + half * 16, qr + kb * 32 + half * 16 + 8);
  }

  v8f o0 = {}, o1 = {}, o2 = {}, o3 = {};
  float mi[8], li[8];
  #pragma unroll
  for (int r = 0; r < 8; ++r) { mi[r] = -1e30f; li[r] = 0.0f; }

  bf16* pw = &pbuf[wave][0];

  for (int mb = 0; mb < 128; ++mb) {
    const int cur = mb & 1, nxt = cur ^ 1;
    if (mb + 1 < 128) {
      const size_t m1 = (size_t)(mb + 1) * 32;
      async_copy_b128(&kbuf[nxt][krow * 64 + koff],
                      Kbh + (m1 + krow) * DHEAD + koff);
      async_copy_b128(&vbuf[nxt][vrow * 32 + voff],
                      Vbh + (size_t)vrow * NTOK + m1 + voff);
      asm volatile("s_wait_asynccnt 0x2" ::: "memory");  // chunk mb landed
    } else {
      asm volatile("s_wait_asynccnt 0x0" ::: "memory");
    }
    __syncthreads();                              // all waves' copies visible

    // ---- S = Q @ K^T over this 32-key chunk (from LDS) ----
    const bf16* kr0 = &kbuf[cur][(size_t)l16 * 64];         // keys 0..15
    const bf16* kr1 = &kbuf[cur][(size_t)(16 + l16) * 64];  // keys 16..31
    v8f s0 = {}, s1 = {};
    #pragma unroll
    for (int kb = 0; kb < 2; ++kb)
      s0 = wmma_bf16(qf[kb], ld_frag16(kr0 + kb * 32 + half * 16), s0);
    #pragma unroll
    for (int kb = 0; kb < 2; ++kb)
      s1 = wmma_bf16(qf[kb], ld_frag16(kr1 + kb * 32 + half * 16), s1);

    // ---- online softmax (reductions within 16-lane halves; rows r+half*8) --
    float alpha[8];
    #pragma unroll
    for (int r = 0; r < 8; ++r) {
      float cm = fmaxf(s0[r], s1[r]);
      #pragma unroll
      for (int m = 1; m < 16; m <<= 1) cm = fmaxf(cm, __shfl_xor(cm, m, 32));
      const float nm = fmaxf(mi[r], cm);
      alpha[r] = __expf(mi[r] - nm);
      mi[r] = nm;
      const float p0 = __expf(s0[r] - nm);
      const float p1 = __expf(s1[r] - nm);
      s0[r] = p0; s1[r] = p1;
      float rs = p0 + p1;
      #pragma unroll
      for (int m = 1; m < 16; m <<= 1) rs += __shfl_xor(rs, m, 32);
      li[r] = li[r] * alpha[r] + rs;
    }
    #pragma unroll
    for (int r = 0; r < 8; ++r) {
      o0[r] *= alpha[r]; o1[r] *= alpha[r]; o2[r] *= alpha[r]; o3[r] *= alpha[r];
    }

    // ---- D-layout -> A-layout transpose of P through per-wave LDS tile ----
    #pragma unroll
    for (int r = 0; r < 8; ++r) {
      const int row = r + half * 8;
      pw[row * 32 + l16]      = (__bf16)s0[r];
      pw[row * 32 + 16 + l16] = (__bf16)s1[r];
    }
    asm volatile("s_wait_dscnt 0" ::: "memory");
    v16bf pf = ld_frag8x2(pw + l16 * 32 + half * 16,
                          pw + l16 * 32 + 8 + half * 16);

    // ---- O += P(16x32) @ V^T(32x64) (V tile from LDS) ----
    const bf16* vr = &vbuf[cur][(size_t)l16 * 32 + half * 16];
    o0 = wmma_bf16(pf, ld_frag16(vr),            o0);
    o1 = wmma_bf16(pf, ld_frag16(vr + 16 * 32),  o1);
    o2 = wmma_bf16(pf, ld_frag16(vr + 32 * 32),  o2);
    o3 = wmma_bf16(pf, ld_frag16(vr + 48 * 32),  o3);

    __syncthreads();   // everyone done reading buf[cur] before it is refilled
  }

  // Normalize and store O as [b][n][hidden] bf16 (B-layout for out-proj).
  #pragma unroll
  for (int r = 0; r < 8; ++r) {
    const float inv = 1.0f / li[r];
    const int n = qt * 16 + r + half * 8;
    bf16* orow = O + ((size_t)b * NTOK + n) * HIDDEN + h * DHEAD + l16;
    orow[0]  = (__bf16)(o0[r] * inv);
    orow[16] = (__bf16)(o1[r] * inv);
    orow[32] = (__bf16)(o2[r] * inv);
    orow[48] = (__bf16)(o3[r] * inv);
  }
}

// ---------------------------------------------------------------------------
// Kernel 3: out = w_out(256x256) @ Oattn(256x4096) + b_out.  Each wave does a
// 16(o) x 64(n) block: the converted w_out A-fragment is reused by 4 WMMAs.
// ---------------------------------------------------------------------------
__global__ __launch_bounds__(256) void out_proj_kernel(
    const bf16* __restrict__ Oin, const float* __restrict__ w_out,
    const float* __restrict__ b_out, float* __restrict__ out) {
  const int wave = threadIdx.x >> 5;
  const int lane = threadIdx.x & 31;
  const int half = lane >> 4;
  const int l16  = lane & 15;
  const int gid  = blockIdx.x * 8 + wave;        // 2*16*64 = 2048 waves
  const int b    = gid / (16 * 64);
  const int rem  = gid % (16 * 64);
  const int ot   = rem / 64;                     // 16 o-tiles
  const int nb   = rem % 64;                     // 64 n-blocks of 64 cols

  const int arow = ot * 16 + l16;
  v8f acc[4] = {{}, {}, {}, {}};
  #pragma unroll
  for (int kb = 0; kb < 8; ++kb) {
    const float* ap = w_out + (size_t)arow * HIDDEN + kb * 32 + half * 16;
    v16bf af = cvt_afrag_f32(ap, ap + 8);
    #pragma unroll
    for (int j = 0; j < 4; ++j) {
      const int n = nb * 64 + j * 16 + l16;
      const bf16* brow = Oin + ((size_t)b * NTOK + n) * HIDDEN;
      acc[j] = wmma_bf16(af, ld_frag16(brow + kb * 32 + half * 16), acc[j]);
    }
  }
  #pragma unroll
  for (int r = 0; r < 8; ++r) {
    const int o = ot * 16 + r + half * 8;
    const float bias = b_out[o];
    #pragma unroll
    for (int j = 0; j < 4; ++j) {
      const int n = nb * 64 + j * 16 + l16;
      out[((size_t)b * HIDDEN + o) * NTOK + n] = acc[j][r] + bias;
    }
  }
}

// ---------------------------------------------------------------------------
extern "C" void kernel_launch(void* const* d_in, const int* in_sizes, int n_in,
                              void* d_out, int out_size, void* d_ws, size_t ws_size,
                              hipStream_t stream) {
  const float* x     = (const float*)d_in[0];
  const float* w_qkv = (const float*)d_in[1];
  const float* w_out = (const float*)d_in[2];
  const float* b_out = (const float*)d_in[3];
  float* out = (float*)d_out;

  const size_t QKV_ELEMS = (size_t)BATCH * HEADS * NTOK * DHEAD;  // 2M elems
  bf16* Q = (bf16*)d_ws;
  bf16* K = Q + QKV_ELEMS;
  bf16* V = K + QKV_ELEMS;
  bf16* O = V + QKV_ELEMS;   // [b][n][hidden], 2M elems (ws total: 16 MB)

  hipLaunchKernelGGL(qkv_proj_kernel, dim3(768), dim3(256), 0, stream,
                     x, w_qkv, Q, K, V);
  hipLaunchKernelGGL(attn_kernel, dim3(256), dim3(256), 0, stream,
                     Q, K, V, O);
  hipLaunchKernelGGL(out_proj_kernel, dim3(256), dim3(256), 0, stream,
                     O, w_out, b_out, out);
}
